// getCostVolume_51719996178539
// MI455X (gfx1250) — matrix operand order repaired and checked by reading, not compiled, not run on hardware
//
#include <hip/hip_runtime.h>

typedef _Float16 half_t;
typedef _Float16 v16h __attribute__((ext_vector_type(16)));
typedef _Float16 v8h  __attribute__((ext_vector_type(8)));
typedef float    v8f  __attribute__((ext_vector_type(8)));

#define BATCH   2
#define NPTS    8192
#define DFEAT   64
#define KNN     16
#define CIN0    131          // 2*D + 3
#define CPAD    160          // layer-0 K padded to multiple of 32
#define KPAD    160          // packed-weight K stride (halves); 320B rows, 16B aligned
#define STRIDE  168          // LDS activation row stride in halves (336B, 16B aligned)
#define COUT    128
#define LEAKY_S 0.1f
#define TILE_Q  4            // query points per MLP block
#define ROWS    (TILE_Q*KNN) // 64 GEMM rows per block

// ---------------------------------------------------------------------------
// One-time prep: pack the 3 MLP weight matrices into f16, K-padded with zeros,
// uniform [layer][o][KPAD] layout so the GEMM B-fragment load is two
// unconditional global_load_b128 per lane.
// ---------------------------------------------------------------------------
__global__ void __launch_bounds__(256)
pack_weights_kernel(const float* __restrict__ w0, const float* __restrict__ w1,
                    const float* __restrict__ w2, half_t* __restrict__ out) {
    int e = blockIdx.x * 256 + threadIdx.x;
    if (e >= 3 * COUT * KPAD) return;
    int layer = e / (COUT * KPAD);
    int rem   = e % (COUT * KPAD);
    int o     = rem / KPAD;
    int k     = rem % KPAD;
    const float* W = (layer == 0) ? w0 : ((layer == 1) ? w1 : w2);
    int cin        = (layer == 0) ? CIN0 : COUT;
    float v = (k < cin) ? W[o * cin + k] : 0.0f;
    out[e] = (half_t)v;
}

// ---------------------------------------------------------------------------
// Brute-force KNN: one thread per query, candidates tiled through LDS,
// in-register sorted top-16 (ascending distance).
// ---------------------------------------------------------------------------
__global__ void __launch_bounds__(256)
knn_kernel(const float* __restrict__ q_xyz,
           const float* __restrict__ db_xyz,
           int* __restrict__ idx_out) {
    __shared__ float sx[256], sy[256], sz[256];
    const int t  = threadIdx.x;
    const int gq = blockIdx.x * 256 + t;       // N divisible by 256 -> one b per block
    const int b  = gq / NPTS;
    const int n  = gq % NPTS;

    const float qx = q_xyz[(b*3+0)*NPTS + n];
    const float qy = q_xyz[(b*3+1)*NPTS + n];
    const float qz = q_xyz[(b*3+2)*NPTS + n];

    float dist[KNN];
    int   idc [KNN];
#pragma unroll
    for (int i = 0; i < KNN; ++i) { dist[i] = 3.0e38f; idc[i] = 0; }

    for (int base = 0; base < NPTS; base += 256) {
        __syncthreads();
        sx[t] = db_xyz[(b*3+0)*NPTS + base + t];
        sy[t] = db_xyz[(b*3+1)*NPTS + base + t];
        sz[t] = db_xyz[(b*3+2)*NPTS + base + t];
        __syncthreads();
        for (int j = 0; j < 256; ++j) {
            float dx = qx - sx[j], dy = qy - sy[j], dz = qz - sz[j];
            float d  = dx*dx + dy*dy + dz*dz;
            if (d < dist[KNN-1]) {
                dist[KNN-1] = d; idc[KNN-1] = base + j;
#pragma unroll
                for (int s = KNN-1; s > 0; --s) {
                    if (dist[s] < dist[s-1]) {
                        float td = dist[s]; dist[s] = dist[s-1]; dist[s-1] = td;
                        int   ti = idc[s];  idc[s]  = idc[s-1];  idc[s-1]  = ti;
                    }
                }
            }
        }
    }
#pragma unroll
    for (int k = 0; k < KNN; ++k)
        idx_out[((size_t)(b*NPTS + n))*KNN + k] = idc[k];
}

// ---------------------------------------------------------------------------
// Fused cost-volume build + 3-layer MLP (WMMA f16) + WeightNet1 + K-reduce.
// One block = 4 query points = 64 rows. 8 waves; wave w owns out-cols [16w,16w+16).
// Weights come from the packed f16 buffer (L2-hot, unconditional b128 loads).
// ---------------------------------------------------------------------------
__global__ void __launch_bounds__(256)
mlp_kernel(const float* __restrict__ xyz1, const float* __restrict__ xyz2,
           const float* __restrict__ p1,   const float* __restrict__ p2,
           const half_t* __restrict__ wpad,
           const float* __restrict__ b0, const float* __restrict__ b1,
           const float* __restrict__ b2,
           const float* __restrict__ wn_w0, const float* __restrict__ wn_b0,
           const float* __restrict__ wn_w1, const float* __restrict__ wn_b1,
           const float* __restrict__ wn_w2, const float* __restrict__ wn_b2,
           const int* __restrict__ idx1, float* __restrict__ p2p) {
    __shared__ __align__(16) half_t bufA[ROWS][STRIDE];
    __shared__ __align__(16) half_t bufB[ROWS][STRIDE];
    __shared__ int   sIdx[ROWS];
    __shared__ float sDir[ROWS][4];
    __shared__ float sH[ROWS][8];

    const int t   = threadIdx.x;
    const int pg0 = blockIdx.x * TILE_Q;       // TILE_Q divides NPTS -> one b per block
    const int b   = pg0 / NPTS;
    const int n0  = pg0 % NPTS;

    if (t < ROWS) {
        int q = t >> 4, k = t & 15;
        sIdx[t] = idx1[((size_t)(b*NPTS + (n0 + q)))*KNN + k];
    }
    __syncthreads();

    // Build cv rows = [p1 | p2(nbr) | dir], f32 -> f16, zero-padded to CPAD.
    for (int e = t; e < ROWS*CPAD; e += 256) {   // 10240/256 = 40, uniform trip count
        int r = e / CPAD, c = e % CPAD;
        int q = r >> 4;
        int n = n0 + q;
        int j = sIdx[r];
        float v;
        if (c < DFEAT)            v = p1[(b*DFEAT + c)*NPTS + n];
        else if (c < 2*DFEAT)     v = p2[(b*DFEAT + (c - DFEAT))*NPTS + j];
        else if (c < CIN0) {
            int ax = c - 2*DFEAT;
            v = xyz2[(b*3 + ax)*NPTS + j] - xyz1[(b*3 + ax)*NPTS + n];
            sDir[r][ax] = v;
        } else                    v = 0.0f;
        bufA[r][c] = (half_t)v;
    }
    __syncthreads();

    const int lane = t & 31;
    const int wv   = t >> 5;          // wave id 0..7 -> output col tile
    const int hi   = lane >> 4;       // lane half (K offset select)
    const int nlo  = lane & 15;
    const int colBase = wv * 16;
    const int o       = colBase + nlo;   // this lane's output channel

    const float* Bl[3] = { b0, b1, b2 };

    half_t (*ain)[STRIDE]  = bufA;
    half_t (*aout)[STRIDE] = bufB;

#pragma unroll
    for (int layer = 0; layer < 3; ++layer) {
        const int ksteps = (layer == 0) ? 5 : 4;
        const half_t* Wrow = wpad + ((size_t)layer*COUT + o)*KPAD;  // lane's W row
        const float  bias  = Bl[layer][o];

        v8f acc[TILE_Q];
#pragma unroll
        for (int r = 0; r < TILE_Q; ++r)
#pragma unroll
            for (int i = 0; i < 8; ++i) acc[r][i] = 0.0f;

#pragma unroll
        for (int ks = 0; ks < ksteps; ++ks) {
            const int kbase = ks * 32;
            // B fragment: 16 consecutive padded-f16 weights -> two b128 loads
            const int kst = kbase + 16*hi;
            v8h blo = *(const v8h*)(Wrow + kst);
            v8h bhi = *(const v8h*)(Wrow + kst + 8);
            v16h bf;
#pragma unroll
            for (int i = 0; i < 8; ++i) { bf[i] = blo[i]; bf[8+i] = bhi[i]; }
#pragma unroll
            for (int r = 0; r < TILE_Q; ++r) {
                const int m  = r*16 + nlo;
                const int k0 = kbase + 16*hi;       // K {0..7}+16*hi
                const int k1 = kbase + 16*hi + 8;   // K {8..15}+16*hi
                v8h alo = *(const v8h*)&ain[m][k0];
                v8h ahi = *(const v8h*)&ain[m][k1];
                v16h af;
#pragma unroll
                for (int i = 0; i < 8; ++i) { af[i] = alo[i]; af[8+i] = ahi[i]; }
                acc[r] = __builtin_amdgcn_wmma_f32_16x16x32_f16(
                    false, af, false, bf, (short)0, acc[r], false, false);
            }
        }
        // Epilogue: bias + leaky relu, D layout -> LDS (m = g + 8*hi)
#pragma unroll
        for (int r = 0; r < TILE_Q; ++r) {
#pragma unroll
            for (int g = 0; g < 8; ++g) {
                int   m = r*16 + g + 8*hi;
                float v = acc[r][g] + bias;
                v = (v > 0.0f) ? v : LEAKY_S * v;
                aout[m][colBase + nlo] = (half_t)v;
            }
        }
        __syncthreads();
        half_t (*tmp)[STRIDE] = ain; ain = aout; aout = tmp;
    }
    // Final activations (cv3, rows x 128) now in `ain`.

    // WeightNet1: 3->8->8 hidden per row (plain ReLU), one thread per row.
    if (t < ROWS) {
        float dx = sDir[t][0], dy = sDir[t][1], dz = sDir[t][2];
        float h0[8];
#pragma unroll
        for (int oo = 0; oo < 8; ++oo) {
            float s = wn_w0[oo*3+0]*dx + wn_w0[oo*3+1]*dy + wn_w0[oo*3+2]*dz + wn_b0[oo];
            h0[oo] = s > 0.0f ? s : 0.0f;
        }
#pragma unroll
        for (int oo = 0; oo < 8; ++oo) {
            float s = wn_b1[oo];
#pragma unroll
            for (int i = 0; i < 8; ++i) s += wn_w1[oo*8+i]*h0[i];
            sH[t][oo] = s > 0.0f ? s : 0.0f;
        }
    }
    __syncthreads();

    // point2patch[q][c] = sum_k relu(W2 . h1[row]) * cv3[row][c]
    {
        const int q  = t >> 6;            // 0..3
        const int c2 = (t & 63) * 2;      // channel pair
        const int n  = n0 + q;
#pragma unroll
        for (int cc = 0; cc < 2; ++cc) {
            int c = c2 + cc;
            float wrow[8];
#pragma unroll
            for (int i = 0; i < 8; ++i) wrow[i] = wn_w2[c*8+i];
            float bb  = wn_b2[c];
            float sum = 0.0f;
#pragma unroll
            for (int k = 0; k < KNN; ++k) {
                int r = q*16 + k;
                float s = bb;
#pragma unroll
                for (int i = 0; i < 8; ++i) s += wrow[i]*sH[r][i];
                s = s > 0.0f ? s : 0.0f;
                sum += s * (float)ain[r][c];
            }
            p2p[((size_t)(b*NPTS + n))*COUT + c] = sum;
        }
    }
}

// ---------------------------------------------------------------------------
// Self-KNN gather + WeightNet2 + K-reduce + transposed store [B,C,N].
// One block per (b,n); thread = output channel.
// ---------------------------------------------------------------------------
__global__ void __launch_bounds__(128)
final_kernel(const float* __restrict__ xyz1, const int* __restrict__ idx2,
             const float* __restrict__ p2p,
             const float* __restrict__ w0, const float* __restrict__ b0,
             const float* __restrict__ w1, const float* __restrict__ b1,
             const float* __restrict__ w2, const float* __restrict__ b2,
             float* __restrict__ outp) {
    __shared__ float sH[KNN][8];
    __shared__ int   sJ[KNN];
    const int t  = threadIdx.x;
    const int pg = blockIdx.x;
    const int b  = pg / NPTS;
    const int n  = pg % NPTS;

    if (t < KNN) {
        int j = idx2[((size_t)(b*NPTS + n))*KNN + t];
        sJ[t] = j;
        float dx = xyz1[(b*3+0)*NPTS + j] - xyz1[(b*3+0)*NPTS + n];
        float dy = xyz1[(b*3+1)*NPTS + j] - xyz1[(b*3+1)*NPTS + n];
        float dz = xyz1[(b*3+2)*NPTS + j] - xyz1[(b*3+2)*NPTS + n];
        float h0[8];
#pragma unroll
        for (int o = 0; o < 8; ++o) {
            float s = w0[o*3+0]*dx + w0[o*3+1]*dy + w0[o*3+2]*dz + b0[o];
            h0[o] = s > 0.0f ? s : 0.0f;
        }
#pragma unroll
        for (int o = 0; o < 8; ++o) {
            float s = b1[o];
#pragma unroll
            for (int i = 0; i < 8; ++i) s += w1[o*8+i]*h0[i];
            sH[t][o] = s > 0.0f ? s : 0.0f;
        }
    }
    __syncthreads();

    const int c = t;
    float wrow[8];
#pragma unroll
    for (int i = 0; i < 8; ++i) wrow[i] = w2[c*8+i];
    const float bb = b2[c];
    float acc = 0.0f;
#pragma unroll
    for (int k = 0; k < KNN; ++k) {
        float s = bb;
#pragma unroll
        for (int i = 0; i < 8; ++i) s += wrow[i]*sH[k][i];
        s = s > 0.0f ? s : 0.0f;
        acc += s * p2p[((size_t)(b*NPTS + sJ[k]))*COUT + c];
    }
    outp[((size_t)(b*COUT + c))*NPTS + n] = acc;   // [B, Cout, N]
}

// ---------------------------------------------------------------------------
extern "C" void kernel_launch(void* const* d_in, const int* in_sizes, int n_in,
                              void* d_out, int out_size, void* d_ws, size_t ws_size,
                              hipStream_t stream) {
    (void)in_sizes; (void)n_in; (void)out_size; (void)ws_size;
    const float* xyz1    = (const float*)d_in[0];
    const float* xyz2    = (const float*)d_in[1];
    const float* points1 = (const float*)d_in[2];
    const float* points2 = (const float*)d_in[3];
    const float* mw0 = (const float*)d_in[4],  *mb0 = (const float*)d_in[5];
    const float* mw1 = (const float*)d_in[6],  *mb1 = (const float*)d_in[7];
    const float* mw2 = (const float*)d_in[8],  *mb2 = (const float*)d_in[9];
    const float* a_w0 = (const float*)d_in[10], *a_b0 = (const float*)d_in[11];
    const float* a_w1 = (const float*)d_in[12], *a_b1 = (const float*)d_in[13];
    const float* a_w2 = (const float*)d_in[14], *a_b2 = (const float*)d_in[15];
    const float* c_w0 = (const float*)d_in[16], *c_b0 = (const float*)d_in[17];
    const float* c_w1 = (const float*)d_in[18], *c_b1 = (const float*)d_in[19];
    const float* c_w2 = (const float*)d_in[20], *c_b2 = (const float*)d_in[21];

    char* ws = (char*)d_ws;
    const size_t idxBytes = (size_t)BATCH*NPTS*KNN*sizeof(int);        // 1 MB
    const size_t p2pBytes = (size_t)BATCH*NPTS*COUT*sizeof(float);     // 8 MB
    int*    idx1 = (int*)ws;
    int*    idx2 = (int*)(ws + idxBytes);
    float*  p2p  = (float*)(ws + 2*idxBytes);
    half_t* wpad = (half_t*)(ws + 2*idxBytes + p2pBytes);              // 120 KB

    // pack weights (3*128*160 = 61440 elems)
    pack_weights_kernel<<<dim3((3*COUT*KPAD + 255)/256), 256, 0, stream>>>(
        mw0, mw1, mw2, wpad);

    const dim3 knnGrid(BATCH*NPTS/256);
    // cross KNN: queries = cloud1, database = cloud2
    knn_kernel<<<knnGrid, 256, 0, stream>>>(xyz1, xyz2, idx1);
    // self KNN on cloud1
    knn_kernel<<<knnGrid, 256, 0, stream>>>(xyz1, xyz1, idx2);

    mlp_kernel<<<dim3(BATCH*NPTS/TILE_Q), 256, 0, stream>>>(
        xyz1, xyz2, points1, points2,
        wpad, mb0, mb1, mb2,
        a_w0, a_b0, a_w1, a_b1, a_w2, a_b2,
        idx1, p2p);

    final_kernel<<<dim3(BATCH*NPTS), 128, 0, stream>>>(
        xyz1, idx2, p2p,
        c_w0, c_b0, c_w1, c_b1, c_w2, c_b2,
        (float*)d_out);
}
